// SNN_Model_LIF_88218628260288
// MI455X (gfx1250) — matrix operand order repaired and checked by reading, not compiled, not run on hardware
//
#include <hip/hip_runtime.h>
#include <hip/hip_bf16.h>

// ---------------------------------------------------------------------------
// SNN LIF model, fused for MI455X (gfx1250).
//   a = x@W1^T + b1          : 16384x1296x512 GEMM -> bf16x3 split WMMA,
//                              W1/x K-chunks async-staged to LDS (double buf)
//   15-step LIF (a constant) : elementwise, in registers
//   head = (s/win)@Wc + bc   : 16384x512x100 GEMM  -> f32 16x16x4 WMMA
// M-tile 64 per block (16 waves) cuts W1 L2 traffic 4x vs M=16.
// ---------------------------------------------------------------------------

typedef __attribute__((ext_vector_type(16))) __bf16 v16bf;
typedef __attribute__((ext_vector_type(8)))  __bf16 v8bf;
typedef __attribute__((ext_vector_type(8)))  float  v8f;
typedef __attribute__((ext_vector_type(2)))  float  v2f;

#define IN_DIM 1296
#define KPAD   1312            // 41 * 32 (zero-padded K for GEMM1)
#define HDIM   512
#define ODIM   50
#define NHEAD  100             // T*K
#define NJPAD  112             // 7 * 16 (padded head dim)
#define BROWS  16384
#define MROWS  64              // batch rows per block
#define SROW   516             // s LDS row stride (floats, anti-conflict pad)

// LDS layout (dynamic), bytes:
//  W1 stage: buf b at b*81920: hi[512 rows x 80B] then lo at +40960
//  x  stage: 163840 + b*9216: 64 rows x 144B (128B payload + pad)
//  s matrix: offset 0 (reused after GEMM1): 64 x 516 floats = 132096B
#define LDS_W1BUF 81920u
#define LDS_XOFF  163840u
#define LDS_XBUF  9216u
#define LDS_TOTAL 182272u

// workspace layout (bytes); total ~2.92 MB
#define OFF_W1HI 0u
#define OFF_W1LO (HDIM * KPAD * 2u)              // 1,343,488
#define OFF_WC   (OFF_W1LO + HDIM * KPAD * 2u)   // 2,686,976
#define OFF_BC   (OFF_WC + HDIM * NJPAD * 4u)    // 2,916,352

// --- prep 1: split W1 (fp32 [512][1296]) into bf16 hi/lo, K padded to 1312 --
__global__ void prep_w1_split(const float* __restrict__ W1,
                              __bf16* __restrict__ hi, __bf16* __restrict__ lo) {
    int idx = blockIdx.x * 256 + threadIdx.x;
    if (idx >= HDIM * KPAD) return;
    int h = idx / KPAD, k = idx - h * KPAD;
    float v = (k < IN_DIM) ? W1[(size_t)h * IN_DIM + k] : 0.0f;
    __bf16 hv = (__bf16)v;
    hi[idx] = hv;
    lo[idx] = (__bf16)(v - (float)hv);
}

// --- prep 2: fused head weights -------------------------------------------
__global__ void prep_heads(const float* __restrict__ W2, const float* __restrict__ b2,
                           const float* __restrict__ Wt, const float* __restrict__ bt,
                           float* __restrict__ Wc, float* __restrict__ bc) {
    int idx = blockIdx.x * 256 + threadIdx.x;
    if (idx < HDIM * NJPAD) {
        int h = idx / NJPAD, j = idx - h * NJPAD;
        float s = 0.0f;
        if (j < NHEAD)
            for (int o = 0; o < ODIM; ++o)
                s += W2[(size_t)o * HDIM + h] * Wt[(size_t)j * ODIM + o];
        Wc[idx] = s;
    }
    if (idx < NJPAD) {
        float s = 0.0f;
        if (idx < NHEAD) {
            for (int o = 0; o < ODIM; ++o) s += b2[o] * Wt[(size_t)idx * ODIM + o];
            s += bt[idx];
        }
        bc[idx] = s;
    }
}

// --- async staging of one 32-wide K chunk into LDS (per thread work) -------
__device__ __forceinline__ void stage_chunk(char* smem, uint32_t smbase, int buf,
                                            int k0, int tid, int b0,
                                            const __bf16* w1hi, const __bf16* w1lo,
                                            const float* x) {
    // W1 hi/lo: thread t copies row h=t, 64B each (4 x async b128).
    const uint32_t gvo = (uint32_t)tid * (KPAD * 2u) + (uint32_t)k0 * 2u;
    const uint32_t lh  = smbase + (uint32_t)buf * LDS_W1BUF + (uint32_t)tid * 80u;
    const uint32_t ll  = lh + 40960u;
    const unsigned long long bh = (unsigned long long)(uintptr_t)w1hi;
    const unsigned long long bl = (unsigned long long)(uintptr_t)w1lo;
    asm volatile(
        "global_load_async_to_lds_b128 %0, %1, %2 offset:0\n\t"
        "global_load_async_to_lds_b128 %0, %1, %2 offset:16\n\t"
        "global_load_async_to_lds_b128 %0, %1, %2 offset:32\n\t"
        "global_load_async_to_lds_b128 %0, %1, %2 offset:48\n\t"
        "global_load_async_to_lds_b128 %3, %1, %4 offset:0\n\t"
        "global_load_async_to_lds_b128 %3, %1, %4 offset:16\n\t"
        "global_load_async_to_lds_b128 %3, %1, %4 offset:32\n\t"
        "global_load_async_to_lds_b128 %3, %1, %4 offset:48"
        :: "v"(lh), "v"(gvo), "s"(bh), "v"(ll), "s"(bl) : "memory");
    // x chunk: thread t copies 16B: row = t>>3, seg = t&7 (zero-fill K tail).
    const int row = tid >> 3, seg = tid & 7;
    const int kk = k0 + seg * 4;
    const uint32_t lxo = LDS_XOFF + (uint32_t)buf * LDS_XBUF
                       + (uint32_t)row * 144u + (uint32_t)seg * 16u;
    if (kk < IN_DIM) {
        const uint32_t gx = ((uint32_t)(b0 + row) * IN_DIM + (uint32_t)kk) * 4u;
        const unsigned long long bx = (unsigned long long)(uintptr_t)x;
        asm volatile("global_load_async_to_lds_b128 %0, %1, %2 offset:0"
                     :: "v"(smbase + lxo), "v"(gx), "s"(bx) : "memory");
    } else {
        float4 z; z.x = 0.f; z.y = 0.f; z.z = 0.f; z.w = 0.f;
        *(float4*)(smem + lxo) = z;
    }
}

__device__ __forceinline__ v16bf join16(v8bf a, v8bf b) {
    return __builtin_shufflevector(a, b, 0,1,2,3,4,5,6,7,8,9,10,11,12,13,14,15);
}

// --- main fused kernel: block = 64 rows of B, 512 threads = 16 waves --------
__global__ __launch_bounds__(512)
void snn_fused(const float* __restrict__ x,
               const __bf16* __restrict__ w1hi, const __bf16* __restrict__ w1lo,
               const float* __restrict__ b1,
               const float* __restrict__ Wc, const float* __restrict__ bc,
               const int* __restrict__ winp, float* __restrict__ out) {
    extern __shared__ __align__(16) char smem[];
    const uint32_t smbase = (uint32_t)(uintptr_t)smem;   // LDS byte offset

    const int tid    = threadIdx.x;
    const int lane   = tid & 31;
    const int wv     = tid >> 5;        // wave 0..15 (uniform per wave)
    const int mg     = wv >> 2;         // M-group 0..3 (16 rows each)
    const int ng     = wv & 3;          // N-group 0..3 (8 h-tiles each)
    const int lane16 = lane & 15;
    const int hiH    = lane >> 4;
    const int b0     = blockIdx.x * MROWS;
    const int win    = *winp;

    // ---------------- GEMM1: a = x @ W1^T (bf16x3 split) -------------------
    // A 16x32 bf16: lanes 0-15 K=0..15, lanes 16-31 K=16..31 (contiguous, as B)
    // B 32x16 bf16: lane n holds K 0..15 (lanes 0-15) / 16..31 contiguously.
    v8f acc[8] = {};

    stage_chunk(smem, smbase, 0, 0, tid, b0, w1hi, w1lo, x);
    asm volatile("s_wait_asynccnt 0x0" ::: "memory");
    __syncthreads();

    int it = 0;
    for (int k0 = 0; k0 < KPAD; k0 += 32, ++it) {
        const int cur = it & 1;
        if (k0 + 32 < KPAD)
            stage_chunk(smem, smbase, cur ^ 1, k0 + 32, tid, b0, w1hi, w1lo, x);

        // A fragment from staged x, split to bf16 hi/lo.
        const uint32_t xb = LDS_XOFF + (uint32_t)cur * LDS_XBUF
                          + (uint32_t)(mg * 16 + lane16) * 144u + (uint32_t)hiH * 64u;
        float4 fa0 = *(const float4*)(smem + xb);
        float4 fa1 = *(const float4*)(smem + xb + 16);
        float4 fa2 = *(const float4*)(smem + xb + 32);
        float4 fa3 = *(const float4*)(smem + xb + 48);
        float af[16] = { fa0.x, fa0.y, fa0.z, fa0.w, fa1.x, fa1.y, fa1.z, fa1.w,
                         fa2.x, fa2.y, fa2.z, fa2.w, fa3.x, fa3.y, fa3.z, fa3.w };
        v16bf ahi, alo;
        #pragma unroll
        for (int j = 0; j < 16; ++j) {
            float v = af[j];
            __bf16 h = (__bf16)v;
            ahi[j] = h;
            alo[j] = (__bf16)(v - (float)h);
        }

        #pragma unroll
        for (int tt = 0; tt < 8; ++tt) {
            const uint32_t bo = (uint32_t)cur * LDS_W1BUF
                              + (uint32_t)((ng * 8 + tt) * 16 + lane16) * 80u
                              + (uint32_t)hiH * 32u;
            v16bf bh = join16(*(const v8bf*)(smem + bo),
                              *(const v8bf*)(smem + bo + 16));
            v16bf blv = join16(*(const v8bf*)(smem + bo + 40960),
                               *(const v8bf*)(smem + bo + 40960 + 16));
            acc[tt] = __builtin_amdgcn_wmma_f32_16x16x32_bf16(
                          false, ahi, false, bh, (short)0, acc[tt], false, false);
            acc[tt] = __builtin_amdgcn_wmma_f32_16x16x32_bf16(
                          false, alo, false, bh, (short)0, acc[tt], false, false);
            acc[tt] = __builtin_amdgcn_wmma_f32_16x16x32_bf16(
                          false, ahi, false, blv, (short)0, acc[tt], false, false);
        }

        asm volatile("s_wait_asynccnt 0x0" ::: "memory");
        __syncthreads();
    }

    // ---------------- LIF recurrence + stage s into LDS --------------------
    float* sS = (float*)smem;                 // reuses staging LDS, 64 x SROW
    const float inv_win = 1.0f / (float)win;
    #pragma unroll
    for (int tt = 0; tt < 8; ++tt) {
        const int h = (ng * 8 + tt) * 16 + lane16;
        const float bias = b1[h];
        float a[8], mem[8], sp[8], sum[8];
        #pragma unroll
        for (int r = 0; r < 8; ++r) {
            a[r] = acc[tt][r] + bias;
            mem[r] = 0.0f; sp[r] = 0.0f; sum[r] = 0.0f;
        }
        for (int t = 0; t < win; ++t) {
            #pragma unroll
            for (int r = 0; r < 8; ++r) {
                mem[r] = mem[r] * 0.2f * (1.0f - sp[r]) + a[r];
                sp[r]  = (mem[r] > 0.3f) ? 1.0f : 0.0f;
                sum[r] += sp[r];
            }
        }
        #pragma unroll
        for (int r = 0; r < 8; ++r)
            sS[(mg * 16 + r + hiH * 8) * SROW + h] = sum[r] * inv_win;
    }
    __syncthreads();

    // ---------------- GEMM2: out = s @ Wc + bc (f32 16x16x4 WMMA) ----------
    // 28 jobs (4 M-groups x 7 j-tiles) over 16 waves; branches wave-uniform.
    #pragma unroll
    for (int jj = 0; jj < 2; ++jj) {
        const int job = wv + jj * 16;
        if (job >= 28) continue;
        const int mg2 = job / 7, jt = job % 7;
        const int col = jt * 16 + lane16;
        v8f o = {};
        for (int kk = 0; kk < HDIM / 4; ++kk) {
            const int ka = kk * 4 + hiH * 2;
            const v2f a2 = *(const v2f*)&sS[(mg2 * 16 + lane16) * SROW + ka];
            v2f bb;
            bb.x = Wc[(size_t)ka * NJPAD + col];
            bb.y = Wc[(size_t)(ka + 1) * NJPAD + col];
            o = __builtin_amdgcn_wmma_f32_16x16x4_f32(
                    false, a2, false, bb, (short)0, o, false, false);
        }
        const int j = col;
        if (j < NHEAD) {
            const float bias = bc[j];
            #pragma unroll
            for (int r = 0; r < 8; ++r)
                out[(size_t)(b0 + mg2 * 16 + r + hiH * 8) * NHEAD + j] = o[r] + bias;
        }
    }
}

// ---------------------------------------------------------------------------
extern "C" void kernel_launch(void* const* d_in, const int* in_sizes, int n_in,
                              void* d_out, int out_size, void* d_ws, size_t ws_size,
                              hipStream_t stream) {
    const float* x  = (const float*)d_in[0];
    const float* W1 = (const float*)d_in[1];
    const float* b1 = (const float*)d_in[2];
    const float* W2 = (const float*)d_in[3];
    const float* b2 = (const float*)d_in[4];
    const float* Wt = (const float*)d_in[5];
    const float* bt = (const float*)d_in[6];
    const int*  win = (const int*)d_in[7];

    char* ws = (char*)d_ws;                  // needs ~2.92 MB of scratch
    __bf16* w1hi = (__bf16*)(ws + OFF_W1HI);
    __bf16* w1lo = (__bf16*)(ws + OFF_W1LO);
    float*  Wc   = (float*)(ws + OFF_WC);
    float*  bc   = (float*)(ws + OFF_BC);

    prep_w1_split<<<(HDIM * KPAD + 255) / 256, 256, 0, stream>>>(W1, w1hi, w1lo);
    prep_heads<<<(HDIM * NJPAD + 255) / 256, 256, 0, stream>>>(W2, b2, Wt, bt, Wc, bc);
    snn_fused<<<BROWS / MROWS, 512, LDS_TOTAL, stream>>>(x, w1hi, w1lo, b1, Wc, bc,
                                                         win, (float*)d_out);
}
